// DecoderLayer_18837726560494
// MI455X (gfx1250) — matrix operand order repaired and checked by reading, not compiled, not run on hardware
//
#include <hip/hip_runtime.h>
#include <hip/hip_bf16.h>
#include <math.h>

#define NNODES 16384
#define NEDGES 131072
#define DIM    256
#define NH     8
#define HDIM   32
#define INTERM 682
#define IPAD   768   // INTERM padded to a multiple of TN/TK (exact: pad cols are zero)

typedef __attribute__((ext_vector_type(16))) __bf16 v16bf;
typedef __attribute__((ext_vector_type(8)))  float  v8f;
typedef __attribute__((ext_vector_type(4)))  unsigned int v4u;
typedef unsigned short u16;

// ---------------- helpers ----------------

__device__ __forceinline__ u16 f32_to_bf16_rne(float f) {
  union { float f; unsigned int u; } c; c.f = f;
  unsigned int x = c.u;
  x += 0x7FFFu + ((x >> 16) & 1u);
  return (u16)(x >> 16);
}

__device__ __forceinline__ unsigned int pack_bf16x2(float a, float b) {
  return (unsigned int)f32_to_bf16_rne(a) | ((unsigned int)f32_to_bf16_rne(b) << 16);
}

__device__ __forceinline__ void atomicMaxFloat(float* addr, float val) {
  // valid when buffer initialized to -inf
  if (val >= 0.0f) atomicMax((int*)addr, __float_as_int(val));
  else             atomicMin((unsigned int*)addr, (unsigned int)__float_as_int(val));
}

// ---------------- WMMA GEMM: C[M,N] = A[M,K] @ B[K,N]. A,B are bf16 (row-major),
// C fp32. Block tile 128x128, K-step 32, 8 waves, double-buffered LDS. Staging is
// a pure copy: global uint4 (b128) loads -> LDS b128 (A) / fragment-packed b16 (B)
// stores; zero conversion VALU in the loop. Fragment reads are ds_load_b128 pairs.
// REQUIREMENTS: M%128==0, N%128==0, K%32==0, lda/ldb %8==0, ldc%4==0. ----------------

#define TM 128
#define TN 128
#define TK 32
#define NT (TN / 16)   // 8 n-tiles per wave

__global__ __launch_bounds__(256) void k_gemm_bf16wmma(
    const u16* __restrict__ A, const u16* __restrict__ B, float* __restrict__ C,
    int M, int Nn, int K, int lda, int ldb, int ldc)
{
  __shared__ u16 lA[2][TM * TK];       // 2 x 8 KB
  __shared__ u16 lB[2][NT * 32 * 16];  // 2 x 8 KB

  const int tid  = threadIdx.x;
  const int wave = tid >> 5;               // 0..7 -> 16-row strip
  const int lane = tid & 31;
  const int r16  = lane & 15;
  const int hsel = lane >> 4;              // half-wave select
  const int rowBase = blockIdx.y * TM;
  const int colBase = blockIdx.x * TN;

  v8f zero = {0,0,0,0,0,0,0,0};
  v8f acc[NT];
  #pragma unroll
  for (int i = 0; i < NT; ++i) acc[i] = zero;

  // per-lane A fragment offsets (contiguous K order, matching B packing):
  // lanes 0-15: row, K [0..7] then [8..15]; lanes 16-31: K [16..23] then [24..31]
  const int arow = wave * 16 + r16;
  const int kb1  = hsel * 16;
  const int kb2  = kb1 + 8;

  uint4 ra[2], rb[2];

  // global -> regs: 2+2 b128 loads per thread, coalesced, no guards
  auto loadTile = [&](int k0) {
    #pragma unroll
    for (int it = 0; it < 2; ++it) {           // A: 128x32 bf16 = 512 x 8-elem chunks
      int idx = tid + it * 256;
      int r  = idx >> 2;
      int kq = (idx & 3) << 3;
      ra[it] = *(const uint4*)(A + (size_t)(rowBase + r) * lda + k0 + kq);
    }
    #pragma unroll
    for (int it = 0; it < 2; ++it) {           // B: 32x128 bf16 = 512 x 8-elem chunks
      int idx = tid + it * 256;
      int kk = idx >> 4;
      int cq = (idx & 15) << 3;
      rb[it] = *(const uint4*)(B + (size_t)(k0 + kk) * ldb + colBase + cq);
    }
  };

  // regs -> LDS: A contiguous b128 stores; B fragment-packed b16 scatter
  auto storeTile = [&](int p) {
    #pragma unroll
    for (int it = 0; it < 2; ++it) {
      int idx = tid + it * 256;
      *(uint4*)(&lA[p][idx << 3]) = ra[it];
    }
    #pragma unroll
    for (int it = 0; it < 2; ++it) {
      int idx = tid + it * 256;
      int kk  = idx >> 4;
      int cq  = (idx & 15) << 3;
      int nt  = cq >> 4;                       // 8 cols stay inside one 16-block
      int i   = kk & 15;
      int lnb = (cq & 15) | (kk & 16);
      const unsigned int* w = (const unsigned int*)&rb[it].x;
      u16* base = &lB[p][0];
      #pragma unroll
      for (int j = 0; j < 4; ++j) {
        unsigned int u = w[j];
        base[(((nt * 32) + lnb + 2 * j + 0) << 4) | i] = (u16)u;
        base[(((nt * 32) + lnb + 2 * j + 1) << 4) | i] = (u16)(u >> 16);
      }
    }
  };

  const int nk = K / TK;
  loadTile(0);
  storeTile(0);
  __syncthreads();

  for (int t = 0; t < nk; ++t) {
    const int cur = t & 1;
    if (t + 1 < nk) loadTile((t + 1) * TK);    // HBM latency hidden under WMMAs

    union { v16bf v; v4u q[2]; } af;
    af.q[0] = *(const v4u*)(&lA[cur][arow * TK + kb1]);
    af.q[1] = *(const v4u*)(&lA[cur][arow * TK + kb2]);

    #pragma unroll
    for (int nt = 0; nt < NT; ++nt) {
      union { v16bf v; v4u q[2]; } bfm;
      const v4u* pb = (const v4u*)(&lB[cur][(nt * 32 + lane) << 4]);
      bfm.q[0] = pb[0];
      bfm.q[1] = pb[1];
      acc[nt] = __builtin_amdgcn_wmma_f32_16x16x32_bf16(
          false, af.v, false, bfm.v, (short)0, acc[nt], false, false);
    }

    if (t + 1 < nk) storeTile((t + 1) & 1);    // write other buffer, one barrier/iter
    __syncthreads();
  }

  // store C (tile-exact: no guards). ISA f32 C/D layout: lane0-15 -> N=lane,
  // VGPR r -> M=r (lanes 0-15) or M=r+8 (lanes 16-31).
  #pragma unroll
  for (int nt = 0; nt < NT; ++nt) {
    int col = colBase + nt * 16 + r16;
    #pragma unroll
    for (int r = 0; r < 8; ++r) {
      int row = rowBase + wave * 16 + hsel * 8 + r;
      C[(size_t)row * ldc + col] = acc[nt][r];
    }
  }
}

// ---------------- conversion / elementwise / edge kernels ----------------

__global__ void k_fill(float* __restrict__ p, float v, int n) {
  int i = blockIdx.x * 256 + threadIdx.x;
  if (i < n) p[i] = v;
}

// fp32 -> bf16, vectorized (n must be a multiple of 4; n4 = n/4)
__global__ void k_cvt_bf16(const float* __restrict__ src, u16* __restrict__ dst, int n4) {
  int i = blockIdx.x * 256 + threadIdx.x;
  if (i >= n4) return;
  float4 v = ((const float4*)src)[i];
  uint2 w;
  w.x = pack_bf16x2(v.x, v.y);
  w.y = pack_bf16x2(v.z, v.w);
  ((uint2*)dst)[i] = w;
}

// zero-padded fp32 -> bf16 copy: dst[r][c] = (r<srcR && c<srcC) ? src : 0
__global__ void k_padcvt(const float* __restrict__ src, u16* __restrict__ dst,
                         int srcR, int srcC, int dstC, int total)
{
  int i = blockIdx.x * 256 + threadIdx.x;
  if (i >= total) return;
  int r = i / dstC, c = i - r * dstC;
  float v = (r < srcR && c < srcC) ? src[(size_t)r * srcC + c] : 0.0f;
  dst[i] = f32_to_bf16_rne(v);
}

// scores[e,h] = scale * sum_d q[dst](h,d) * k[src](h,d) * (ea ? ea[e,h,d] : 1)
// then atomicMax into m[dst,h]
__global__ void k_edge_score(const float* __restrict__ qbuf, int qrow, int qmul, int qoff,
                             const float* __restrict__ kbuf, int krow, int kmul, int koff,
                             const float* __restrict__ ea,
                             const int* __restrict__ src, const int* __restrict__ dst,
                             float* __restrict__ scores, float* __restrict__ m,
                             int nE, float scale)
{
  int t = blockIdx.x * 256 + threadIdx.x;
  if (t >= nE * NH) return;
  int e = t >> 3, h = t & 7;
  int si = src[e], di = dst[e];
  const float* qp = qbuf + (size_t)di * qrow + h * HDIM * qmul + qoff;
  const float* kp = kbuf + (size_t)si * krow + h * HDIM * kmul + koff;
  float acc = 0.0f;
  if (ea) {
    const float* ep = ea + (size_t)e * DIM + h * HDIM;
    #pragma unroll 4
    for (int i = 0; i < HDIM; ++i) acc += qp[i * qmul] * kp[i * kmul] * ep[i];
  } else {
    #pragma unroll 4
    for (int i = 0; i < HDIM; ++i) acc += qp[i * qmul] * kp[i * kmul];
  }
  acc *= scale;
  scores[(size_t)e * NH + h] = acc;
  atomicMaxFloat(&m[(size_t)di * NH + h], acc);
}

// scores[e,h] = exp(scores - m[dst,h]);  s[dst,h] += scores
__global__ void k_edge_exp(float* __restrict__ scores, const float* __restrict__ m,
                           float* __restrict__ s, const int* __restrict__ dst, int nE)
{
  int t = blockIdx.x * 256 + threadIdx.x;
  if (t >= nE * NH) return;
  int e = t >> 3, h = t & 7;
  int di = dst[e];
  float ex = expf(scores[t] - m[(size_t)di * NH + h]);
  scores[t] = ex;
  atomicAdd(&s[(size_t)di * NH + h], ex);
}

// agg[dst, d] += (scores[e,h]/s[dst,h]) * V[src](h,hd)   (thread per (e,d))
__global__ void k_edge_agg(const float* __restrict__ scores, const float* __restrict__ s,
                           const float* __restrict__ vbuf, int vrow, int vmul, int voff,
                           const int* __restrict__ src, const int* __restrict__ dst,
                           float* __restrict__ agg, int nE)
{
  int t = blockIdx.x * 256 + threadIdx.x;
  if (t >= nE * DIM) return;
  int e = t >> 8, d = t & 255;
  int h = d >> 5;
  int si = src[e], di = dst[e];
  float atn = scores[(size_t)e * NH + h] / s[(size_t)di * NH + h];
  float vv  = vbuf[(size_t)si * vrow + d * vmul + voff];
  atomicAdd(&agg[(size_t)di * DIM + d], atn * vv);
}

// out = rms_norm(a + b) * g  (one 256-thread block per row, D=256)
// optional bf16 side output for feeding downstream GEMMs
__global__ __launch_bounds__(256) void k_res_rmsnorm(
    const float* __restrict__ a, const float* __restrict__ b,
    const float* __restrict__ g, float* __restrict__ out, u16* __restrict__ outbf)
{
  __shared__ float red[256];
  int row = blockIdx.x, t = threadIdx.x;
  float x = a[(size_t)row * DIM + t] + b[(size_t)row * DIM + t];
  red[t] = x * x;
  __syncthreads();
  for (int sft = 128; sft > 0; sft >>= 1) {
    if (t < sft) red[t] += red[t + sft];
    __syncthreads();
  }
  float norm = sqrtf(red[0]) * (1.0f / 16.0f);   // ||x|| * D^-0.5, D=256
  float inv  = 1.0f / fmaxf(norm, 1e-8f);
  float y = x * inv * g[t];
  out[(size_t)row * DIM + t] = y;
  if (outbf) outbf[(size_t)row * DIM + t] = f32_to_bf16_rne(y);
}

// act_bf = bf16( h1 * sigmoid(h1) * h2 )  -- written directly in bf16 for the GEMM
__global__ void k_swiglu_bf(const float* __restrict__ h1, const float* __restrict__ h2,
                            u16* __restrict__ act, int n)
{
  int i = blockIdx.x * 256 + threadIdx.x;
  if (i >= n) return;
  float h = h1[i];
  float sig = 1.0f / (1.0f + expf(-h));
  act[i] = f32_to_bf16_rne(h * sig * h2[i]);
}

// ---------------- launch ----------------

extern "C" void kernel_launch(void* const* d_in, const int* in_sizes, int n_in,
                              void* d_out, int out_size, void* d_ws, size_t ws_size,
                              hipStream_t stream) {
  (void)in_sizes; (void)n_in; (void)out_size; (void)ws_size;

  const float* root   = (const float*)d_in[0];
  const float* node   = (const float*)d_in[1];
  const float* fringe = (const float*)d_in[2];
  const int* ntr_idx  = (const int*)d_in[3];
  const int* rtr_idx  = (const int*)d_in[4];
  const int* rtf_idx  = (const int*)d_in[5];
  const float* redge  = (const float*)d_in[6];
  const float* ntr_wq   = (const float*)d_in[7];
  const float* ntr_wkv  = (const float*)d_in[8];
  const float* ntr_wout = (const float*)d_in[9];
  const float* ntr_g    = (const float*)d_in[10];
  const float* rtr_wqkv = (const float*)d_in[11];
  const float* rtr_wout = (const float*)d_in[12];
  const float* rtr_g    = (const float*)d_in[13];
  const float* ffn_win  = (const float*)d_in[14];
  const float* ffn_v    = (const float*)d_in[15];
  const float* ffn_wout = (const float*)d_in[16];
  const float* ffn_g    = (const float*)d_in[17];
  const float* rtf_wq   = (const float*)d_in[18];
  const float* rtf_wkv  = (const float*)d_in[19];
  const float* rtf_wout = (const float*)d_in[20];

  float* outp = (float*)d_out;

  // workspace carve: fp32 region then bf16 region (~245 MB total)
  float* ws = (float*)d_ws;
  size_t off = 0;
  float* BIG    = ws + off; off += (size_t)NNODES * (2 * IPAD); // h1+h2 / q+kv / qkv
  float* x1     = ws + off; off += (size_t)NNODES * DIM;
  float* x2     = ws + off; off += (size_t)NNODES * DIM;
  float* tmp    = ws + off; off += (size_t)NNODES * DIM;
  float* scores = ws + off; off += (size_t)NEDGES * NH;
  float* mbuf   = ws + off; off += (size_t)NNODES * NH;
  float* sbuf   = ws + off; off += (size_t)NNODES * NH;
  float* agg    = ws + off; off += (size_t)NNODES * DIM;

  u16* wsb = (u16*)(ws + off);
  size_t uo = 0;
  u16* root_bf   = wsb + uo; uo += (size_t)NNODES * DIM;
  u16* node_bf   = wsb + uo; uo += (size_t)NNODES * DIM;
  u16* fringe_bf = wsb + uo; uo += (size_t)NNODES * DIM;
  u16* x1_bf     = wsb + uo; uo += (size_t)NNODES * DIM;
  u16* agg_bf    = wsb + uo; uo += (size_t)NNODES * DIM;
  u16* act_bf    = wsb + uo; uo += (size_t)NNODES * IPAD;
  u16* bwq1   = wsb + uo; uo += (size_t)DIM * DIM;
  u16* bwkv1  = wsb + uo; uo += (size_t)DIM * 2 * DIM;
  u16* bwo1   = wsb + uo; uo += (size_t)DIM * DIM;
  u16* bwqkv  = wsb + uo; uo += (size_t)DIM * 3 * DIM;
  u16* bwo2   = wsb + uo; uo += (size_t)DIM * DIM;
  u16* bpwin  = wsb + uo; uo += (size_t)DIM * IPAD;
  u16* bpv    = wsb + uo; uo += (size_t)DIM * IPAD;
  u16* bpwout = wsb + uo; uo += (size_t)IPAD * DIM;
  u16* bwq4   = wsb + uo; uo += (size_t)DIM * DIM;
  u16* bwkv4  = wsb + uo; uo += (size_t)DIM * 2 * DIM;
  u16* bwo4   = wsb + uo; uo += (size_t)DIM * DIM;

  auto cvt = [&](const float* s, u16* d, int n) {
    int n4 = n >> 2;
    k_cvt_bf16<<<(n4 + 255) / 256, 256, 0, stream>>>(s, d, n4);
  };

  auto gemm = [&](const u16* A, const u16* B, float* C, int M, int Nn, int K) {
    dim3 grid(Nn / TN, M / TM);  // all calls tile-exact by construction
    k_gemm_bf16wmma<<<grid, 256, 0, stream>>>(A, B, C, M, Nn, K, K, Nn, Nn);
  };

  const int EH  = NEDGES * NH;
  const int ED  = NEDGES * DIM;
  const int NH8 = NNODES * NH;
  const int ND  = NNODES * DIM;

  auto attention = [&](const float* qbuf, int qrow, int qmul,
                       const float* kvbuf, int krow, int kmul, int koff, int voff,
                       const float* ea, const int* eidx, float scale, float* aggOut) {
    const int* src = eidx;
    const int* dst = eidx + NEDGES;
    k_fill<<<(NH8 + 255) / 256, 256, 0, stream>>>(mbuf, -INFINITY, NH8);
    k_fill<<<(NH8 + 255) / 256, 256, 0, stream>>>(sbuf, 0.0f, NH8);
    k_fill<<<(ND + 255) / 256, 256, 0, stream>>>(aggOut, 0.0f, ND);
    k_edge_score<<<(EH + 255) / 256, 256, 0, stream>>>(
        qbuf, qrow, qmul, 0, kvbuf, krow, kmul, koff, ea, src, dst,
        scores, mbuf, NEDGES, scale);
    k_edge_exp<<<(EH + 255) / 256, 256, 0, stream>>>(scores, mbuf, sbuf, dst, NEDGES);
    k_edge_agg<<<(ED + 255) / 256, 256, 0, stream>>>(
        scores, sbuf, kvbuf, krow, kmul, voff, src, dst, aggOut, NEDGES);
  };

  const float inv_sqrt_hd = 0.17677669529663689f; // 1/sqrt(32)

  // ---- one-time bf16 conversions (activations read by multiple GEMMs + weights) ----
  cvt(root,   root_bf,   NNODES * DIM);
  cvt(node,   node_bf,   NNODES * DIM);
  cvt(fringe, fringe_bf, NNODES * DIM);
  cvt(ntr_wq,   bwq1,  DIM * DIM);
  cvt(ntr_wkv,  bwkv1, DIM * 2 * DIM);
  cvt(ntr_wout, bwo1,  DIM * DIM);
  cvt(rtr_wqkv, bwqkv, DIM * 3 * DIM);
  cvt(rtr_wout, bwo2,  DIM * DIM);
  cvt(rtf_wq,   bwq4,  DIM * DIM);
  cvt(rtf_wkv,  bwkv4, DIM * 2 * DIM);
  cvt(rtf_wout, bwo4,  DIM * DIM);
  {
    int tw = DIM * IPAD;
    k_padcvt<<<(tw + 255) / 256, 256, 0, stream>>>(ffn_win, bpwin, DIM, INTERM, IPAD, tw);
    k_padcvt<<<(tw + 255) / 256, 256, 0, stream>>>(ffn_v,   bpv,   DIM, INTERM, IPAD, tw);
    int to = IPAD * DIM;
    k_padcvt<<<(to + 255) / 256, 256, 0, stream>>>(ffn_wout, bpwout, INTERM, DIM, DIM, to);
  }

  // ---- nodes_to_root CrossMHA (w_out hoisted past segment_sum by linearity) ----
  float* q  = BIG;
  float* kv = BIG + (size_t)NNODES * DIM;
  gemm(root_bf, bwq1,  q,  NNODES, DIM,     DIM);
  gemm(node_bf, bwkv1, kv, NNODES, 2 * DIM, DIM);
  attention(q, DIM, 1, kv, 2 * DIM, 2, /*koff=*/0, /*voff=*/1,
            nullptr, ntr_idx, inv_sqrt_hd, agg);
  cvt(agg, agg_bf, NNODES * DIM);
  gemm(agg_bf, bwo1, tmp, NNODES, DIM, DIM);
  k_res_rmsnorm<<<NNODES, 256, 0, stream>>>(root, tmp, ntr_g, x1, x1_bf);

  // ---- roots_to_root SelfMHA ----
  float* qkv = BIG;
  gemm(x1_bf, bwqkv, qkv, NNODES, 3 * DIM, DIM);
  attention(qkv, 3 * DIM, 3, qkv, 3 * DIM, 3, /*koff=*/1, /*voff=*/2,
            redge, rtr_idx, 1.0f, agg);
  cvt(agg, agg_bf, NNODES * DIM);
  gemm(agg_bf, bwo2, tmp, NNODES, DIM, DIM);
  k_res_rmsnorm<<<NNODES, 256, 0, stream>>>(x1, tmp, rtr_g, x2, (u16*)nullptr);

  // ---- SwiGLU FFN on root_features (INTERM padded 682 -> 768, exact) ----
  float* h1 = BIG;
  float* h2 = BIG + (size_t)NNODES * IPAD;
  gemm(root_bf, bpwin, h1, NNODES, IPAD, DIM);
  gemm(root_bf, bpv,   h2, NNODES, IPAD, DIM);
  {
    int n = NNODES * IPAD;  // pad cols: h1==0 -> act==0, contributes nothing
    k_swiglu_bf<<<(n + 255) / 256, 256, 0, stream>>>(h1, h2, act_bf, n);
  }
  gemm(act_bf, bpwout, tmp, NNODES, DIM, IPAD);
  k_res_rmsnorm<<<NNODES, 256, 0, stream>>>(tmp, x2, ffn_g, outp, (u16*)nullptr); // output 0

  // ---- root_to_fringe CrossMHA ----
  gemm(fringe_bf, bwq4,  q,  NNODES, DIM,     DIM);
  gemm(root_bf,   bwkv4, kv, NNODES, 2 * DIM, DIM);
  attention(q, DIM, 1, kv, 2 * DIM, 2, /*koff=*/0, /*voff=*/1,
            nullptr, rtf_idx, inv_sqrt_hd, agg);
  cvt(agg, agg_bf, NNODES * DIM);
  gemm(agg_bf, bwo4, outp + (size_t)NNODES * DIM, NNODES, DIM, DIM); // output 1
}
